// ScaledDotProductAttention_79937931313827
// MI455X (gfx1250) — compile-verified, matrix operand order checked by reading, not run on hardware
//
#include <hip/hip_runtime.h>
#include <hip/hip_bf16.h>

// ---------------------------------------------------------------------------
// Multi-head attention (B=16, Nq=Nk=784, d_model=512, H=8, dk=dv=64)
// f16 WMMA (v_wmma_f32_16x16x32_f16) + flash softmax + async-to-LDS staging.
// ---------------------------------------------------------------------------

typedef __attribute__((ext_vector_type(16))) _Float16 v16h;
typedef __attribute__((ext_vector_type(8)))  _Float16 v8h;
typedef __attribute__((ext_vector_type(8)))  float    v8f;

static_assert(sizeof(v16h) == 32, "v16h");
static_assert(sizeof(v8h) == 16, "v8h");

union hv { v16h v; v8h h[2]; };

#define DM    512
#define NSEQ  784
#define NKP   800          // keys padded to multiple of 32 (pad rows zeroed)
#define HEADS 8
#define BATCH 16

#define WMMA16(A, B, C) \
  __builtin_amdgcn_wmma_f32_16x16x32_f16(false, (A), false, (B), (short)0, (C), false, false)

// pack 8 f32 -> 8 f16 with v_cvt_pk_rz_f16_f32
__device__ __forceinline__ v8h pk8(float4 a, float4 b) {
  v8h r;
  auto p0 = __builtin_amdgcn_cvt_pkrtz(a.x, a.y);
  auto p1 = __builtin_amdgcn_cvt_pkrtz(a.z, a.w);
  auto p2 = __builtin_amdgcn_cvt_pkrtz(b.x, b.y);
  auto p3 = __builtin_amdgcn_cvt_pkrtz(b.z, b.w);
  __builtin_memcpy((char*)&r + 0,  &p0, 4);
  __builtin_memcpy((char*)&r + 4,  &p1, 4);
  __builtin_memcpy((char*)&r + 8,  &p2, 4);
  __builtin_memcpy((char*)&r + 12, &p3, 4);
  return r;
}

// CDNA5 async global->LDS copy, 16B per lane, tracked by ASYNCcnt.
__device__ __forceinline__ void async_ld_b128(void* lds, const void* g) {
  unsigned l = (unsigned)(unsigned long long)lds;          // LDS byte offset
  unsigned long long ga = (unsigned long long)g;           // 64-bit global addr
  asm volatile("global_load_async_to_lds_b128 %0, %1, off"
               :: "v"(l), "v"(ga) : "memory");
}
__device__ __forceinline__ void wait_async0() {
  asm volatile("s_wait_asynccnt 0x0" ::: "memory");
}

// ---------------------------------------------------------------------------
// Kernel 1: fused QKV projection.  X[12544,512]f32 @ W[512,512]f32 + b.
// grid = (196 Mtiles, 8 Ntiles(=head), 3 which), block = 128 (4 waves).
// Q stored f16 [b,h,784,64]; K stored f16 [b,h,800,64] (pad rows pre-zeroed);
// V stored f16 transposed [b,h,64,800] (pad cols pre-zeroed).
// ---------------------------------------------------------------------------
__global__ __launch_bounds__(128)
void qkv_proj_kernel(const float* __restrict__ Xq, const float* __restrict__ Xk,
                     const float* __restrict__ Xv,
                     const float* __restrict__ Wq, const float* __restrict__ Wk,
                     const float* __restrict__ Wv,
                     const float* __restrict__ Bq, const float* __restrict__ Bk,
                     const float* __restrict__ Bv,
                     _Float16* __restrict__ qo, _Float16* __restrict__ ko,
                     _Float16* __restrict__ vto)
{
  const int which = blockIdx.z;
  const float* X  = (which == 0) ? Xq : (which == 1) ? Xk : Xv;
  const float* W  = (which == 0) ? Wq : (which == 1) ? Wk : Wv;
  const float* Bs = (which == 0) ? Bq : (which == 1) ? Bk : Bv;

  __shared__ __align__(16) _Float16 As[64 * 32];   // [row][k]
  __shared__ __align__(16) _Float16 Bts[64 * 32];  // [col][k]  (W^T tile)

  const int t    = threadIdx.x;
  const int lane = t & 31;
  const int wave = t >> 5;
  const int m0   = blockIdx.x * 64;
  const int h    = blockIdx.y;
  const int cb   = h * 64;

  const int mlane = lane & 15;
  const int klo   = (lane < 16) ? 0 : 8;

  v8f acc[4] = {};

  for (int k0 = 0; k0 < DM; k0 += 32) {
    __syncthreads();
    // stage A: 64 rows x 32 k (f32 -> f16), packed converts + 16B LDS stores
    {
      const int row = t >> 1;
      const int kc  = (t & 1) * 16;
      const float4* src = (const float4*)(X + (size_t)(m0 + row) * DM + k0 + kc);
      v8h* dst = (v8h*)&As[row * 32 + kc];
      dst[0] = pk8(src[0], src[1]);
      dst[1] = pk8(src[2], src[3]);
    }
    // stage B^T: pack (k, k+1) pairs -> 4B LDS stores. thread: k-pair + 8 cols
    {
      const int kk = (t >> 3) * 2;       // 0,2,..,30
      const int c0 = (t & 7) * 8;        // 0..56
      const float4* s0 = (const float4*)(W + (size_t)(k0 + kk) * DM + cb + c0);
      const float4* s1 = (const float4*)(W + (size_t)(k0 + kk + 1) * DM + cb + c0);
      const float4 a0 = s0[0], a1 = s0[1], b0 = s1[0], b1 = s1[1];
      const float fa[8] = {a0.x, a0.y, a0.z, a0.w, a1.x, a1.y, a1.z, a1.w};
      const float fb[8] = {b0.x, b0.y, b0.z, b0.w, b1.x, b1.y, b1.z, b1.w};
#pragma unroll
      for (int i = 0; i < 8; ++i) {
        auto p = __builtin_amdgcn_cvt_pkrtz(fa[i], fb[i]);
        __builtin_memcpy((void*)&Bts[(c0 + i) * 32 + kk], &p, 4);
      }
    }
    __syncthreads();

    v16h a;
    {
      hv u;
      const _Float16* ar = &As[(wave * 16 + mlane) * 32];
      u.h[0] = *(const v8h*)(ar + klo);
      u.h[1] = *(const v8h*)(ar + klo + 16);
      a = u.v;
    }
#pragma unroll
    for (int tt = 0; tt < 4; ++tt) {
      hv u;
      const _Float16* br = &Bts[(tt * 16 + mlane) * 32];
      u.h[0] = *(const v8h*)(br + klo);
      u.h[1] = *(const v8h*)(br + klo + 16);
      acc[tt] = WMMA16(a, u.v, acc[tt]);
    }
  }

  // epilogue: rows of one lane-half are 8 consecutive n in one batch
  // (m0 + wave*16 + {0|8} is a multiple of 8, and 784 % 8 == 0)
  const int rowg0 = m0 + wave * 16 + ((lane < 16) ? 0 : 8);
  const int b0    = rowg0 / NSEQ;
  const int n0    = rowg0 - b0 * NSEQ;
  const size_t bh = (size_t)b0 * HEADS + h;

#pragma unroll
  for (int tt = 0; tt < 4; ++tt) {
    const int d = tt * 16 + mlane;
    const float bias = Bs[cb + d];
    if (which == 2) {
      // V^T: one 16B store of 8 consecutive n
      const float4 lo = make_float4(acc[tt][0] + bias, acc[tt][1] + bias,
                                    acc[tt][2] + bias, acc[tt][3] + bias);
      const float4 hi = make_float4(acc[tt][4] + bias, acc[tt][5] + bias,
                                    acc[tt][6] + bias, acc[tt][7] + bias);
      *(v8h*)(vto + (bh * 64 + d) * NKP + n0) = pk8(lo, hi);
    } else {
#pragma unroll
      for (int r = 0; r < 8; ++r) {
        const _Float16 val = (_Float16)(acc[tt][r] + bias);
        if (which == 0) qo[(bh * NSEQ + (n0 + r)) * 64 + d] = val;
        else            ko[(bh * NKP  + (n0 + r)) * 64 + d] = val;
      }
    }
  }
}

// ---------------------------------------------------------------------------
// Kernel 2: flash attention with additive bias.
// grid = (13 qblocks, 8 heads, 16 batch), block = 128 (4 waves x 16 q-rows).
// K/V tiles staged with global_load_async_to_lds_b128 (pre-padded buffers).
// ---------------------------------------------------------------------------
__global__ __launch_bounds__(128)
void attn_kernel(const _Float16* __restrict__ q, const _Float16* __restrict__ k,
                 const _Float16* __restrict__ vt, const float* __restrict__ corr,
                 _Float16* __restrict__ ctx)
{
  const int qb = blockIdx.x;
  const int h  = blockIdx.y;
  const int b  = blockIdx.z;
  const int t = threadIdx.x, lane = t & 31, wave = t >> 5;
  const int qbase = qb * 64;

  __shared__ __align__(16) _Float16 Ks[32 * 64];     // [key][d]
  __shared__ __align__(16) _Float16 Vts[64 * 32];    // [d][key]
  __shared__ __align__(16) float    Cs[64 * 32];     // [qrow][key]
  __shared__ __align__(16) _Float16 Ps[4][16 * 32];  // per-wave prob tile

  const size_t bh = (size_t)b * HEADS + h;
  const _Float16* qp = q  + bh * NSEQ * 64;
  const _Float16* kp = k  + bh * NKP * 64;
  const _Float16* vp = vt + bh * 64 * NKP;
  const float*    cp = corr + (size_t)b * NSEQ * NSEQ;

  const int mlane = lane & 15;
  const int klo   = (lane < 16) ? 0 : 8;

  // this wave's Q A-operands (16 rows x 64 d = two 16x32 operands)
  int qrow = qbase + wave * 16 + mlane;
  if (qrow > NSEQ - 1) qrow = NSEQ - 1;
  v16h aq[2];
  {
    const _Float16* qr = qp + (size_t)qrow * 64;
#pragma unroll
    for (int c = 0; c < 2; ++c) {
      hv u;
      u.h[0] = *(const v8h*)(qr + 32 * c + klo);
      u.h[1] = *(const v8h*)(qr + 32 * c + klo + 16);
      aq[c] = u.v;
    }
  }

  float mrun[8], lrun[8];
#pragma unroll
  for (int r = 0; r < 8; ++r) { mrun[r] = -1e30f; lrun[r] = 0.0f; }
  v8f cacc[4] = {};

  for (int k0 = 0; k0 < NKP; k0 += 32) {
    __syncthreads();
    // ---- async stage K tile: 4KB contiguous (rows k0..k0+31, padded buffer)
    {
      const _Float16* gk = kp + (size_t)k0 * 64;
      async_ld_b128(&Ks[(size_t)t * 8],         gk + (size_t)t * 8);
      async_ld_b128(&Ks[(size_t)(t + 128) * 8], gk + (size_t)(t + 128) * 8);
    }
    // ---- async stage V^T tile: 64 rows x 32 keys (row stride NKP)
    {
      const int r0 = t >> 2,         c0 = (t & 3) * 8;
      const int r1 = (t + 128) >> 2, c1 = (t & 3) * 8;
      async_ld_b128(&Vts[r0 * 32 + c0], vp + (size_t)r0 * NKP + k0 + c0);
      async_ld_b128(&Vts[r1 * 32 + c1], vp + (size_t)r1 * NKP + k0 + c1);
    }
    // ---- stage self_corr tile (16-float chunks align with 784 exactly)
    {
      const int r_ = t >> 1, c0 = (t & 1) * 16;
      int qg = qbase + r_;
      if (qg > NSEQ - 1) qg = NSEQ - 1;
      float4* dst = (float4*)&Cs[r_ * 32 + c0];
      if (k0 + c0 < NSEQ) {
        const float4* src = (const float4*)(cp + (size_t)qg * NSEQ + k0 + c0);
        dst[0] = src[0]; dst[1] = src[1]; dst[2] = src[2]; dst[3] = src[3];
      } else {
        const float4 z = make_float4(0.f, 0.f, 0.f, 0.f);
        dst[0] = z; dst[1] = z; dst[2] = z; dst[3] = z;
      }
    }
    if (k0 + 32 < NSEQ)
      __builtin_prefetch(kp + (size_t)(k0 + 32) * 64 + t * 4, 0, 1);
    wait_async0();
    __syncthreads();

    // ---- S = Q @ K^T  (two 16x16 tiles, K-dim 64 = 2 WMMAs each)
    v8f s[2];
#pragma unroll
    for (int nt = 0; nt < 2; ++nt) {
      const int col = nt * 16 + mlane;
      const _Float16* kr0 = &Ks[col * 64];
      v8f sc = {};
      hv u;
      u.h[0] = *(const v8h*)(kr0 + klo);
      u.h[1] = *(const v8h*)(kr0 + klo + 16);
      sc = WMMA16(aq[0], u.v, sc);
      u.h[0] = *(const v8h*)(kr0 + 32 + klo);
      u.h[1] = *(const v8h*)(kr0 + 32 + klo + 16);
      sc = WMMA16(aq[1], u.v, sc);
      s[nt] = sc;
    }

    // ---- scale, additive bias, OOB mask
#pragma unroll
    for (int nt = 0; nt < 2; ++nt) {
      const int col = nt * 16 + mlane;
      const bool oob = (k0 + col) >= NSEQ;
#pragma unroll
      for (int r = 0; r < 8; ++r) {
        const int mr = (lane < 16) ? r : (8 + r);
        float v = s[nt][r] * 0.125f + Cs[(wave * 16 + mr) * 32 + col];
        if (oob) v = -1e30f;
        s[nt][r] = v;
      }
    }

    // ---- online softmax (rows live in 16-lane groups; XOR 1/2/4/8 reduces)
#pragma unroll
    for (int r = 0; r < 8; ++r) {
      float mx = fmaxf(s[0][r], s[1][r]);
      mx = fmaxf(mx, __shfl_xor(mx, 1, 32));
      mx = fmaxf(mx, __shfl_xor(mx, 2, 32));
      mx = fmaxf(mx, __shfl_xor(mx, 4, 32));
      mx = fmaxf(mx, __shfl_xor(mx, 8, 32));
      const float mnew = fmaxf(mrun[r], mx);
      const float cf = __expf(mrun[r] - mnew);
      const float p0 = __expf(s[0][r] - mnew);
      const float p1 = __expf(s[1][r] - mnew);
      s[0][r] = p0;
      s[1][r] = p1;
      float rs = p0 + p1;
      rs += __shfl_xor(rs, 1, 32);
      rs += __shfl_xor(rs, 2, 32);
      rs += __shfl_xor(rs, 4, 32);
      rs += __shfl_xor(rs, 8, 32);
      lrun[r] = lrun[r] * cf + rs;
      mrun[r] = mnew;
#pragma unroll
      for (int tt = 0; tt < 4; ++tt) cacc[tt][r] *= cf;
    }

    // ---- round-trip P through LDS to re-layout into an A operand
#pragma unroll
    for (int nt = 0; nt < 2; ++nt) {
#pragma unroll
      for (int r = 0; r < 8; ++r) {
        const int mr = (lane < 16) ? r : (8 + r);
        Ps[wave][mr * 32 + nt * 16 + mlane] = (_Float16)s[nt][r];
      }
    }
    __syncthreads();

    v16h ap;
    {
      hv u;
      const _Float16* pr = &Ps[wave][mlane * 32];
      u.h[0] = *(const v8h*)(pr + klo);
      u.h[1] = *(const v8h*)(pr + klo + 16);
      ap = u.v;
    }
    // ---- ctx += P @ V   (4 d-tiles of 16, K-dim = 32 keys)
#pragma unroll
    for (int tt = 0; tt < 4; ++tt) {
      hv u;
      const _Float16* vr = &Vts[(tt * 16 + mlane) * 32];
      u.h[0] = *(const v8h*)(vr + klo);
      u.h[1] = *(const v8h*)(vr + klo + 16);
      cacc[tt] = WMMA16(ap, u.v, cacc[tt]);
    }
  }

  // ---- normalize and store ctx f16 [b, n, h*64]
#pragma unroll
  for (int r = 0; r < 8; ++r) {
    const float inv = 1.0f / lrun[r];
    const int mr = (lane < 16) ? r : (8 + r);
    const int qg = qbase + wave * 16 + mr;
    if (qg < NSEQ) {
#pragma unroll
      for (int tt = 0; tt < 4; ++tt) {
        const int d = tt * 16 + mlane;
        ctx[((size_t)b * NSEQ + qg) * (HEADS * 64) + h * 64 + d] =
            (_Float16)(cacc[tt][r] * inv);
      }
    }
  }
}

// ---------------------------------------------------------------------------
// Kernel 3: output projection + transpose.  ctx[12544,512]f16 @ Wo + bo
// -> out[b, c, n] f32.  A tile staged with async-to-LDS copies.
// ---------------------------------------------------------------------------
__global__ __launch_bounds__(128)
void out_proj_kernel(const _Float16* __restrict__ ctxin,
                     const float* __restrict__ Wo, const float* __restrict__ bo,
                     float* __restrict__ out)
{
  __shared__ __align__(16) _Float16 As[64 * 32];
  __shared__ __align__(16) _Float16 Bts[64 * 32];

  const int t = threadIdx.x, lane = t & 31, wave = t >> 5;
  const int m0 = blockIdx.x * 64;
  const int cb = blockIdx.y * 64;

  const int mlane = lane & 15;
  const int klo   = (lane < 16) ? 0 : 8;

  v8f acc[4] = {};

  for (int k0 = 0; k0 < DM; k0 += 32) {
    __syncthreads();
    // async stage A (pure f16 copy): rows m0..m0+63, 32 k each
    {
      const int r0 = t >> 2,         c0 = (t & 3) * 8;
      const int r1 = (t + 128) >> 2;
      async_ld_b128(&As[r0 * 32 + c0], ctxin + (size_t)(m0 + r0) * DM + k0 + c0);
      async_ld_b128(&As[r1 * 32 + c0], ctxin + (size_t)(m0 + r1) * DM + k0 + c0);
    }
    // stage Wo^T: pack (k, k+1) pairs -> 4B LDS stores
    {
      const int kk = (t >> 3) * 2;
      const int c0 = (t & 7) * 8;
      const float4* s0 = (const float4*)(Wo + (size_t)(k0 + kk) * DM + cb + c0);
      const float4* s1 = (const float4*)(Wo + (size_t)(k0 + kk + 1) * DM + cb + c0);
      const float4 a0 = s0[0], a1 = s0[1], b0 = s1[0], b1 = s1[1];
      const float fa[8] = {a0.x, a0.y, a0.z, a0.w, a1.x, a1.y, a1.z, a1.w};
      const float fb[8] = {b0.x, b0.y, b0.z, b0.w, b1.x, b1.y, b1.z, b1.w};
#pragma unroll
      for (int i = 0; i < 8; ++i) {
        auto p = __builtin_amdgcn_cvt_pkrtz(fa[i], fb[i]);
        __builtin_memcpy((void*)&Bts[(c0 + i) * 32 + kk], &p, 4);
      }
    }
    wait_async0();
    __syncthreads();

    v16h a;
    {
      hv u;
      const _Float16* ar = &As[(wave * 16 + mlane) * 32];
      u.h[0] = *(const v8h*)(ar + klo);
      u.h[1] = *(const v8h*)(ar + klo + 16);
      a = u.v;
    }
#pragma unroll
    for (int tt = 0; tt < 4; ++tt) {
      hv u;
      const _Float16* br = &Bts[(tt * 16 + mlane) * 32];
      u.h[0] = *(const v8h*)(br + klo);
      u.h[1] = *(const v8h*)(br + klo + 16);
      acc[tt] = WMMA16(a, u.v, acc[tt]);
    }
  }

  // epilogue: 8 consecutive n per lane-half -> two aligned float4 stores
  const int rowg0 = m0 + wave * 16 + ((lane < 16) ? 0 : 8);
  const int b0    = rowg0 / NSEQ;
  const int n0    = rowg0 - b0 * NSEQ;

#pragma unroll
  for (int tt = 0; tt < 4; ++tt) {
    const int c = cb + tt * 16 + mlane;
    const float bias = bo[c];
    float* dst = out + ((size_t)b0 * DM + c) * NSEQ + n0;
    const float4 lo = make_float4(acc[tt][0] + bias, acc[tt][1] + bias,
                                  acc[tt][2] + bias, acc[tt][3] + bias);
    const float4 hi = make_float4(acc[tt][4] + bias, acc[tt][5] + bias,
                                  acc[tt][6] + bias, acc[tt][7] + bias);
    ((float4*)dst)[0] = lo;
    ((float4*)dst)[1] = hi;
  }
}

// ---------------------------------------------------------------------------
extern "C" void kernel_launch(void* const* d_in, const int* in_sizes, int n_in,
                              void* d_out, int out_size, void* d_ws, size_t ws_size,
                              hipStream_t stream) {
  const float* queries   = (const float*)d_in[0];
  const float* keys      = (const float*)d_in[1];
  const float* values    = (const float*)d_in[2];
  const float* self_corr = (const float*)d_in[3];
  const float* Wq = (const float*)d_in[4];
  const float* bq = (const float*)d_in[5];
  const float* Wk = (const float*)d_in[6];
  const float* bk = (const float*)d_in[7];
  const float* Wv = (const float*)d_in[8];
  const float* bv = (const float*)d_in[9];
  const float* Wo = (const float*)d_in[10];
  const float* bo = (const float*)d_in[11];
  float* out = (float*)d_out;

  // workspace layout: q | k(padded) | v^T(padded) | ctx
  const size_t seg_q  = (size_t)BATCH * HEADS * NSEQ * 64;  // f16 elems
  const size_t seg_kp = (size_t)BATCH * HEADS * NKP * 64;
  _Float16* q_ws   = (_Float16*)d_ws;
  _Float16* k_ws   = q_ws + seg_q;
  _Float16* vt_ws  = k_ws + seg_kp;
  _Float16* ctx_ws = vt_ws + seg_kp;

  // zero K / V^T pad regions (whole segments; capture-safe async memset)
  hipMemsetAsync(k_ws, 0, 2 * seg_kp * sizeof(_Float16), stream);

  const int mtiles  = (BATCH * NSEQ) / 64;  // 196
  const int ntiles  = DM / 64;              // 8
  const int qblocks = (NSEQ + 63) / 64;     // 13

  qkv_proj_kernel<<<dim3(mtiles, ntiles, 3), 128, 0, stream>>>(
      queries, keys, values, Wq, Wk, Wv, bq, bk, bv, q_ws, k_ws, vt_ws);

  attn_kernel<<<dim3(qblocks, HEADS, BATCH), 128, 0, stream>>>(
      q_ws, k_ws, vt_ws, self_corr, ctx_ws);

  out_proj_kernel<<<dim3(mtiles, ntiles), 128, 0, stream>>>(
      ctx_ws, Wo, bo, out);
}
